// BlockCausalTransformer_884763263652
// MI455X (gfx1250) — matrix operand order repaired and checked by reading, not compiled, not run on hardware
//
#include <hip/hip_runtime.h>
#include <hip/hip_bf16.h>
#include <math.h>

// ---------------------------------------------------------------------------
// Block-causal transformer forward for MI455X (gfx1250), wave32 + WMMA bf16.
// GEMMs: v_wmma_f32_16x16x32_bf16 with double-buffered async global->LDS
// copies (global_load_async_to_lds_b128 / s_wait_asynccnt).
// ---------------------------------------------------------------------------

#define SEQ    2048
#define DIMSZ  1024
#define HEADS  16
#define DHEAD  64
#define QKV_N  3072
#define FF_HID 2730
#define FF2    5460           // 2*FF_HID
#define FF2P   5504           // padded to mult of 64 (GEMM N, no edge guards)
#define FFHP   2752           // FF_HID padded to mult of 32 (GEMM K)
#define MIXN   64             // mix head-count padded to mult of 64
#define RMSEPS 1.1920929e-07f
#define ATT_SCALE 0.125f      // 64^-0.5

typedef __attribute__((ext_vector_type(16))) __bf16 v16bf;
typedef __attribute__((ext_vector_type(8)))  float  v8f;

struct __align__(16) U4 { unsigned int x, y, z, w; };
union Frag { U4 q[2]; v16bf v; };

__device__ __forceinline__ unsigned short f2bf(float f) {
  unsigned int u = __float_as_uint(f);
  u += 0x7FFFu + ((u >> 16) & 1u);          // round-to-nearest-even
  return (unsigned short)(u >> 16);
}

// flat shared-space pointer -> wave-relative LDS byte offset (addr[31:0])
__device__ __forceinline__ unsigned lds_off(const void* p) {
  return (unsigned)(size_t)p;
}

// async copy of 16 bytes global -> LDS (no VGPR staging, ASYNCcnt-tracked)
__device__ __forceinline__ void async_cp16(unsigned lds, const void* g) {
  asm volatile("global_load_async_to_lds_b128 %0, %1, off"
               :: "v"(lds), "v"((unsigned long long)(size_t)g)
               : "memory");
}

// ---------------------------------------------------------------------------
// GEMM: C[M][Np] (f32) = A[M][Kp] (bf16 row-major) x Bt[Np][Kp] (bf16,
// pre-transposed).  Kp % 32 == 0, M % 64 == 0, Np % 64 == 0 (pre-padded).
// Block = 256 threads = 8 waves; 64x64 tile; each wave -> 16x32 output.
// Double-buffered LDS fed by async copies.
// ---------------------------------------------------------------------------
#define LDSW 40   // u16 row stride for LDS tiles (32 data + 8 pad)

__global__ __launch_bounds__(256)
void gemm_bf16_wmma(const unsigned short* __restrict__ A,
                    const unsigned short* __restrict__ Bt,
                    float* __restrict__ C,
                    int M, int Kp, int Np)
{
  __shared__ __align__(16) unsigned short As[2][64 * LDSW];
  __shared__ __align__(16) unsigned short Bs[2][64 * LDSW];

  const int tid  = threadIdx.x;
  const int wave = tid >> 5;
  const int lane = tid & 31;
  const int m0 = blockIdx.x * 64;
  const int n0 = blockIdx.y * 64;
  const int tr  = wave >> 1;            // row tile 0..3
  const int tc0 = (wave & 1) * 2;       // col tile base {0,2}

  v8f acc0 = {};
  v8f acc1 = {};

  const int lrow   = tid >> 2;          // 0..63
  const int lchunk = tid & 3;           // 16B chunk of a 32-wide bf16 row

  const unsigned short* gA = A  + (size_t)(m0 + lrow) * Kp + lchunk * 8;
  const unsigned short* gB = Bt + (size_t)(n0 + lrow) * Kp + lchunk * 8;
  const unsigned ldsA0 = lds_off(&As[0][lrow * LDSW + lchunk * 8]);
  const unsigned ldsB0 = lds_off(&Bs[0][lrow * LDSW + lchunk * 8]);
  const unsigned bufStride = lds_off(&As[1][0]) - lds_off(&As[0][0]);

  // prologue: stage K-step 0 into buffer 0
  async_cp16(ldsA0, gA);
  async_cp16(ldsB0, gB);

  int buf = 0;
  for (int k0 = 0; k0 < Kp; k0 += 32, buf ^= 1) {
    const bool last = (k0 + 32 >= Kp);
    if (!last) {
      const unsigned nb = (unsigned)(buf ^ 1) * bufStride;
      async_cp16(ldsA0 + nb, gA + k0 + 32);
      async_cp16(ldsB0 + nb, gB + k0 + 32);
      asm volatile("s_wait_asynccnt 0x2" ::: "memory"); // current buf resident
    } else {
      asm volatile("s_wait_asynccnt 0x0" ::: "memory");
    }
    __syncthreads();   // all waves' async copies of this buffer are done

    Frag a, b0, b1;
    const int arow  = tr * 16 + (lane & 15);
    const int khalf = (lane >> 4) * 16;          // A: K = khalf..khalf+15 contiguous
    a.q[0] = *(const U4*)(&As[buf][arow * LDSW + khalf]);
    a.q[1] = *(const U4*)(&As[buf][arow * LDSW + khalf + 8]);

    const int kb  = (lane >> 4) * 16;            // B: K = kb..kb+15 contiguous
    const int bn0 = tc0 * 16 + (lane & 15);
    const int bn1 = bn0 + 16;
    b0.q[0] = *(const U4*)(&Bs[buf][bn0 * LDSW + kb]);
    b0.q[1] = *(const U4*)(&Bs[buf][bn0 * LDSW + kb + 8]);
    b1.q[0] = *(const U4*)(&Bs[buf][bn1 * LDSW + kb]);
    b1.q[1] = *(const U4*)(&Bs[buf][bn1 * LDSW + kb + 8]);

    acc0 = __builtin_amdgcn_wmma_f32_16x16x32_bf16(false, a.v, false, b0.v,
                                                   (short)0, acc0, false, false);
    acc1 = __builtin_amdgcn_wmma_f32_16x16x32_bf16(false, a.v, false, b1.v,
                                                   (short)0, acc1, false, false);
    __syncthreads();   // buffer may be overwritten next iteration
  }

  const int mbase = m0 + tr * 16 + (lane >> 4) * 8;
  const int nc0   = n0 + tc0 * 16 + (lane & 15);
#pragma unroll
  for (int r = 0; r < 8; ++r) {
    const size_t m = (size_t)(mbase + r);
    C[m * Np + nc0]      = acc0[r];
    C[m * Np + nc0 + 16] = acc1[r];
  }
}

// ---------------------------------------------------------------------------
// Flash-style block-causal attention. One block per (q-block of 128, head).
// 8 waves; wave w owns 16 query rows. Key loop covers blocks 0..qb fully
// (block-causal at 128 granularity => no elementwise masking needed).
// Q,K: bf16 [H][SEQ][64] (rows contiguous).  VT: bf16 [H][64][SEQ] (V^T).
// O:  bf16 [SEQ][H*64].
// ---------------------------------------------------------------------------
__global__ __launch_bounds__(256)
void attn_wmma(const unsigned short* __restrict__ Q,
               const unsigned short* __restrict__ K,
               const unsigned short* __restrict__ VT,
               unsigned short* __restrict__ O)
{
  __shared__ __align__(16) unsigned short Pw[8][16 * 32];  // per-wave P tile

  const int qb   = blockIdx.x;
  const int h    = blockIdx.y;
  const int wave = threadIdx.x >> 5;
  const int lane = threadIdx.x & 31;
  const int qrow0 = qb * 128 + wave * 16;

  const unsigned short* qp = Q  + (size_t)h * SEQ * DHEAD;
  const unsigned short* kp = K  + (size_t)h * SEQ * DHEAD;
  const unsigned short* vp = VT + (size_t)h * DHEAD * SEQ;

  // Q fragments held in registers for the whole kernel (K-dim = head dim).
  Frag qf0, qf1;
  {
    const unsigned short* row = qp + (size_t)(qrow0 + (lane & 15)) * DHEAD;
    const int khalf = (lane >> 4) * 16;
    qf0.q[0] = *(const U4*)(row + khalf);
    qf0.q[1] = *(const U4*)(row + khalf + 8);
    qf1.q[0] = *(const U4*)(row + 32 + khalf);
    qf1.q[1] = *(const U4*)(row + 32 + khalf + 8);
  }

  v8f o0 = {}, o1 = {}, o2 = {}, o3 = {};
  float mr[8], lr[8];
#pragma unroll
  for (int r = 0; r < 8; ++r) { mr[r] = -3.0e38f; lr[r] = 0.f; }

  const int nkeys = (qb + 1) * 128;
  for (int kt = 0; kt < nkeys; kt += 32) {
    // ---- S = Q . K^T : two 16x16 key sub-tiles ----
    Frag kf00, kf01, kf10, kf11;
    {
      const int kb = (lane >> 4) * 16;
      const unsigned short* r0 = kp + (size_t)(kt      + (lane & 15)) * DHEAD;
      const unsigned short* r1 = kp + (size_t)(kt + 16 + (lane & 15)) * DHEAD;
      kf00.q[0] = *(const U4*)(r0 + kb);      kf00.q[1] = *(const U4*)(r0 + kb + 8);
      kf01.q[0] = *(const U4*)(r0 + 32 + kb); kf01.q[1] = *(const U4*)(r0 + 32 + kb + 8);
      kf10.q[0] = *(const U4*)(r1 + kb);      kf10.q[1] = *(const U4*)(r1 + kb + 8);
      kf11.q[0] = *(const U4*)(r1 + 32 + kb); kf11.q[1] = *(const U4*)(r1 + 32 + kb + 8);
    }
    v8f s0 = {}, s1 = {};
    s0 = __builtin_amdgcn_wmma_f32_16x16x32_bf16(false, qf0.v, false, kf00.v, (short)0, s0, false, false);
    s0 = __builtin_amdgcn_wmma_f32_16x16x32_bf16(false, qf1.v, false, kf01.v, (short)0, s0, false, false);
    s1 = __builtin_amdgcn_wmma_f32_16x16x32_bf16(false, qf0.v, false, kf10.v, (short)0, s1, false, false);
    s1 = __builtin_amdgcn_wmma_f32_16x16x32_bf16(false, qf1.v, false, kf11.v, (short)0, s1, false, false);

    // ---- online softmax (each row lives across 16 lanes of a half-wave) ----
#pragma unroll
    for (int r = 0; r < 8; ++r) {
      const float sv0 = s0[r] * ATT_SCALE;
      const float sv1 = s1[r] * ATT_SCALE;
      float t = fmaxf(sv0, sv1);
      for (int off = 1; off < 16; off <<= 1) t = fmaxf(t, __shfl_xor(t, off, 16));
      const float nm   = fmaxf(mr[r], t);
      const float corr = __expf(mr[r] - nm);
      const float p0   = __expf(sv0 - nm);
      const float p1   = __expf(sv1 - nm);
      float su = p0 + p1;
      for (int off = 1; off < 16; off <<= 1) su += __shfl_xor(su, off, 16);
      lr[r] = lr[r] * corr + su;
      mr[r] = nm;
      o0[r] *= corr; o1[r] *= corr; o2[r] *= corr; o3[r] *= corr;

      // P (C-layout) -> LDS, bf16 (same wave reads it back: in-order DS ops)
      const int prow = r + 8 * (lane >> 4);
      Pw[wave][prow * 32 +      (lane & 15)] = f2bf(p0);
      Pw[wave][prow * 32 + 16 + (lane & 15)] = f2bf(p1);
    }

    // ---- O += P . V  (P re-read in A-layout; V fragments from V^T) ----
    Frag pf;
    {
      const int m     = lane & 15;
      const int khalf = (lane >> 4) * 16;
      pf.q[0] = *(const U4*)(&Pw[wave][m * 32 + khalf]);
      pf.q[1] = *(const U4*)(&Pw[wave][m * 32 + khalf + 8]);
    }
    const int kbase = kt + (lane >> 4) * 16;
#pragma unroll
    for (int dt = 0; dt < 4; ++dt) {
      Frag vf;
      const unsigned short* vr = vp + (size_t)(dt * 16 + (lane & 15)) * SEQ + kbase;
      vf.q[0] = *(const U4*)(vr);
      vf.q[1] = *(const U4*)(vr + 8);
      v8f* oacc = (dt == 0) ? &o0 : (dt == 1) ? &o1 : (dt == 2) ? &o2 : &o3;
      *oacc = __builtin_amdgcn_wmma_f32_16x16x32_bf16(false, pf.v, false, vf.v,
                                                      (short)0, *oacc, false, false);
    }
  }

  // ---- normalize + store ----
#pragma unroll
  for (int r = 0; r < 8; ++r) {
    const int m   = qrow0 + r + 8 * (lane >> 4);
    const float inv = 1.f / lr[r];
    unsigned short* orow = O + (size_t)m * (HEADS * DHEAD) + h * DHEAD + (lane & 15);
    orow[0]  = f2bf(o0[r] * inv);
    orow[16] = f2bf(o1[r] * inv);
    orow[32] = f2bf(o2[r] * inv);
    orow[48] = f2bf(o3[r] * inv);
  }
}

// ---------------------------------------------------------------------------
// Elementwise / prep kernels
// ---------------------------------------------------------------------------

// weight convert + transpose + zero-pad: out[Np][Kp] bf16 <- in[K][N] f32
__global__ void conv_wT(const float* __restrict__ in, unsigned short* __restrict__ out,
                        int K, int N, int Kp, int Np)
{
  const int idx = blockIdx.x * 256 + threadIdx.x;
  if (idx >= Np * Kp) return;
  const int k = idx % Kp, n = idx / Kp;
  const float v = (n < N && k < K) ? in[(size_t)k * N + n] : 0.f;
  out[idx] = f2bf(v);
}

__global__ __launch_bounds__(256)
void rmsnorm_bf16(const float* __restrict__ x, const float* __restrict__ w,
                  unsigned short* __restrict__ out)
{
  __shared__ float red[256];
  const int n = blockIdx.x;
  const float* xr = x + (size_t)n * DIMSZ;
  float s = 0.f;
  for (int j = threadIdx.x; j < DIMSZ; j += 256) { const float v = xr[j]; s += v * v; }
  red[threadIdx.x] = s; __syncthreads();
  for (int st = 128; st > 0; st >>= 1) {
    if (threadIdx.x < st) red[threadIdx.x] += red[threadIdx.x + st];
    __syncthreads();
  }
  const float inv = rsqrtf(red[0] / (float)DIMSZ + RMSEPS);
  for (int j = threadIdx.x; j < DIMSZ; j += 256)
    out[(size_t)n * DIMSZ + j] = f2bf(xr[j] * inv * w[j]);
}

__global__ __launch_bounds__(256)
void rmsnorm_f32(const float* __restrict__ x, const float* __restrict__ w,
                 float* __restrict__ out)
{
  __shared__ float red[256];
  const int n = blockIdx.x;
  const float* xr = x + (size_t)n * DIMSZ;
  float s = 0.f;
  for (int j = threadIdx.x; j < DIMSZ; j += 256) { const float v = xr[j]; s += v * v; }
  red[threadIdx.x] = s; __syncthreads();
  for (int st = 128; st > 0; st >>= 1) {
    if (threadIdx.x < st) red[threadIdx.x] += red[threadIdx.x + st];
    __syncthreads();
  }
  const float inv = rsqrtf(red[0] / (float)DIMSZ + RMSEPS);
  for (int j = threadIdx.x; j < DIMSZ; j += 256)
    out[(size_t)n * DIMSZ + j] = xr[j] * inv * w[j];
}

// interleaved RoPE on q,k from the qkv GEMM output; one thread per (n,h,pair)
__global__ void rope_kernel(const float* __restrict__ qkv,
                            unsigned short* __restrict__ qr,
                            unsigned short* __restrict__ kr)
{
  const int idx = blockIdx.x * 256 + threadIdx.x;   // SEQ*HEADS*32 exact
  const int p = idx & 31;
  const int h = (idx >> 5) & 15;
  const int n = idx >> 9;
  const float fr = __expf(-(float)(2 * p) * (9.210340372f / 64.f)); // 10000^(-2p/64)
  float sn, cs;
  __sincosf((float)n * fr, &sn, &cs);
  const float* base = qkv + (size_t)n * QKV_N + h * DHEAD + 2 * p;
  const float q0 = base[0],    q1 = base[1];
  const float k0 = base[1024], k1 = base[1025];
  const size_t o = ((size_t)h * SEQ + n) * DHEAD + 2 * p;
  qr[o]     = f2bf(q0 * cs - q1 * sn);
  qr[o + 1] = f2bf(q1 * cs + q0 * sn);
  kr[o]     = f2bf(k0 * cs - k1 * sn);
  kr[o + 1] = f2bf(k1 * cs + k0 * sn);
}

// v path: save first_v (layer 0) or sigmoid-lerp toward first_v, then
// transpose to V^T bf16 [H][64][SEQ]. One thread per (n,h,d).
__global__ void vprep_kernel(const float* __restrict__ qkv,
                             const float* __restrict__ mixc,   // [SEQ][MIXN]
                             const float* __restrict__ bmix,   // [16]
                             float* __restrict__ firstv,       // [H][SEQ][64]
                             unsigned short* __restrict__ vt,  // [H][64][SEQ]
                             int use_mix)
{
  const int idx = blockIdx.x * 256 + threadIdx.x;   // SEQ*HEADS*64 exact
  const int d = idx & 63;
  const int h = (idx >> 6) & 15;
  const int n = idx >> 10;
  float v = qkv[(size_t)n * QKV_N + 2048 + h * DHEAD + d];
  const size_t fvi = ((size_t)h * SEQ + n) * DHEAD + d;
  if (use_mix) {
    const float z  = mixc[(size_t)n * MIXN + h] + bmix[h];
    const float mx = 1.f / (1.f + __expf(-z));
    const float fv = firstv[fvi];
    v = v + mx * (fv - v);
  } else {
    firstv[fvi] = v;
  }
  vt[((size_t)h * DHEAD + d) * SEQ + n] = f2bf(v);
}

// SwiGLU: hid = (a+ba) * gelu_exact(g+bg), bf16, zero-padded to FFHP cols.
__global__ void swiglu_kernel(const float* __restrict__ cff,   // [SEQ][FF2P]
                              const float* __restrict__ bias,  // [FF2]
                              unsigned short* __restrict__ hid)// [SEQ][FFHP]
{
  const int idx = blockIdx.x * 256 + threadIdx.x;
  if (idx >= SEQ * FFHP) return;
  const int j = idx % FFHP;
  const int n = idx / FFHP;
  float v = 0.f;
  if (j < FF_HID) {
    const float a = cff[(size_t)n * FF2P + j]          + bias[j];
    const float g = cff[(size_t)n * FF2P + FF_HID + j] + bias[FF_HID + j];
    v = a * (0.5f * g * (1.f + erff(g * 0.70710678f)));
  }
  hid[idx] = f2bf(v);
}

__global__ void add_res(float* __restrict__ x, const float* __restrict__ c)
{
  const int idx = blockIdx.x * 256 + threadIdx.x;   // SEQ*DIM exact
  x[idx] += c[idx];
}

__global__ void add_res_bias(float* __restrict__ x, const float* __restrict__ c,
                             const float* __restrict__ b)
{
  const int idx = blockIdx.x * 256 + threadIdx.x;   // SEQ*DIM exact
  x[idx] += c[idx] + b[idx & (DIMSZ - 1)];
}

// ---------------------------------------------------------------------------
// Host-side orchestration
// ---------------------------------------------------------------------------
extern "C" void kernel_launch(void* const* d_in, const int* in_sizes, int n_in,
                              void* d_out, int out_size, void* d_ws, size_t ws_size,
                              hipStream_t stream)
{
  const float* tokens       = (const float*)d_in[0];
  const float* attn_norm_w  = (const float*)d_in[1];
  const float* w_qkv        = (const float*)d_in[2];
  const float* w_attn_out   = (const float*)d_in[3];
  const float* w_mix        = (const float*)d_in[4];
  const float* b_mix        = (const float*)d_in[5];
  const float* ff_norm_w    = (const float*)d_in[6];
  const float* w_ff_in      = (const float*)d_in[7];
  const float* b_ff_in      = (const float*)d_in[8];
  const float* w_ff_out     = (const float*)d_in[9];
  const float* b_ff_out     = (const float*)d_in[10];
  const float* final_norm_w = (const float*)d_in[11];

  char* base = (char*)d_ws;
  size_t off = 0;
  auto alloc = [&](size_t bytes) -> char* {
    char* p = base + off;
    off += (bytes + 255) & ~(size_t)255;
    return p;
  };

  float*          X    = (float*)         alloc((size_t)SEQ * DIMSZ * 4);
  unsigned short* H    = (unsigned short*)alloc((size_t)SEQ * DIMSZ * 2);
  float*          BIG  = (float*)         alloc((size_t)SEQ * FF2P * 4);   // all GEMM outputs
  unsigned short* QR   = (unsigned short*)alloc((size_t)HEADS * SEQ * DHEAD * 2);
  unsigned short* KR   = (unsigned short*)alloc((size_t)HEADS * SEQ * DHEAD * 2);
  unsigned short* VT   = (unsigned short*)alloc((size_t)HEADS * SEQ * DHEAD * 2);
  float*          FV   = (float*)         alloc((size_t)HEADS * SEQ * DHEAD * 4);
  float*          MIXC = (float*)         alloc((size_t)SEQ * MIXN * 4);
  unsigned short* OATT = (unsigned short*)alloc((size_t)SEQ * DIMSZ * 2);
  unsigned short* HID  = (unsigned short*)alloc((size_t)SEQ * FFHP * 2);
  unsigned short* WQKV = (unsigned short*)alloc((size_t)QKV_N * DIMSZ * 2);
  unsigned short* WAO  = (unsigned short*)alloc((size_t)DIMSZ * DIMSZ * 2);
  unsigned short* WMIX = (unsigned short*)alloc((size_t)MIXN * DIMSZ * 2);
  unsigned short* WFFI = (unsigned short*)alloc((size_t)FF2P * DIMSZ * 2);
  unsigned short* WFFO = (unsigned short*)alloc((size_t)DIMSZ * FFHP * 2);
  if (off > ws_size) return;   // scratch too small: bail

  hipMemcpyAsync(X, tokens, (size_t)SEQ * DIMSZ * sizeof(float),
                 hipMemcpyDeviceToDevice, stream);

  const dim3 blk(256);
  auto cdiv = [](int a, int b) { return (a + b - 1) / b; };

  for (int i = 0; i < 4; ++i) {
    // --- per-layer weight convert/transposes (bf16, zero-padded) ---
    conv_wT<<<cdiv(QKV_N * DIMSZ, 256), blk, 0, stream>>>(
        w_qkv + (size_t)i * DIMSZ * QKV_N, WQKV, DIMSZ, QKV_N, DIMSZ, QKV_N);
    conv_wT<<<cdiv(DIMSZ * DIMSZ, 256), blk, 0, stream>>>(
        w_attn_out + (size_t)i * DIMSZ * DIMSZ, WAO, DIMSZ, DIMSZ, DIMSZ, DIMSZ);
    conv_wT<<<cdiv(MIXN * DIMSZ, 256), blk, 0, stream>>>(
        w_mix + (size_t)i * DIMSZ * 16, WMIX, DIMSZ, 16, DIMSZ, MIXN);
    conv_wT<<<cdiv(FF2P * DIMSZ, 256), blk, 0, stream>>>(
        w_ff_in + (size_t)i * DIMSZ * FF2, WFFI, DIMSZ, FF2, DIMSZ, FF2P);
    conv_wT<<<cdiv(DIMSZ * FFHP, 256), blk, 0, stream>>>(
        w_ff_out + (size_t)i * FF_HID * DIMSZ, WFFO, FF_HID, DIMSZ, FFHP, DIMSZ);

    // --- attention branch ---
    rmsnorm_bf16<<<SEQ, blk, 0, stream>>>(X, attn_norm_w + (size_t)i * DIMSZ, H);

    gemm_bf16_wmma<<<dim3(SEQ / 64, QKV_N / 64), blk, 0, stream>>>(
        H, WQKV, BIG, SEQ, DIMSZ, QKV_N);
    if (i > 0)
      gemm_bf16_wmma<<<dim3(SEQ / 64, MIXN / 64), blk, 0, stream>>>(
          H, WMIX, MIXC, SEQ, DIMSZ, MIXN);

    rope_kernel<<<(SEQ * HEADS * 32) / 256, blk, 0, stream>>>(BIG, QR, KR);
    vprep_kernel<<<(SEQ * HEADS * DHEAD) / 256, blk, 0, stream>>>(
        BIG, MIXC, b_mix + (size_t)i * 16, FV, VT, i > 0 ? 1 : 0);

    attn_wmma<<<dim3(SEQ / 128, HEADS), blk, 0, stream>>>(QR, KR, VT, OATT);

    gemm_bf16_wmma<<<dim3(SEQ / 64, DIMSZ / 64), blk, 0, stream>>>(
        OATT, WAO, BIG, SEQ, DIMSZ, DIMSZ);
    add_res<<<(SEQ * DIMSZ) / 256, blk, 0, stream>>>(X, BIG);

    // --- SwiGLU feedforward ---
    rmsnorm_bf16<<<SEQ, blk, 0, stream>>>(X, ff_norm_w + (size_t)i * DIMSZ, H);
    gemm_bf16_wmma<<<dim3(SEQ / 64, FF2P / 64), blk, 0, stream>>>(
        H, WFFI, BIG, SEQ, DIMSZ, FF2P);
    swiglu_kernel<<<cdiv(SEQ * FFHP, 256), blk, 0, stream>>>(
        BIG, b_ff_in + (size_t)i * FF2, HID);
    gemm_bf16_wmma<<<dim3(SEQ / 64, DIMSZ / 64), blk, 0, stream>>>(
        HID, WFFO, BIG, SEQ, FFHP, DIMSZ);
    add_res_bias<<<(SEQ * DIMSZ) / 256, blk, 0, stream>>>(
        X, BIG, b_ff_out + (size_t)i * DIMSZ);
  }

  rmsnorm_f32<<<SEQ, blk, 0, stream>>>(X, final_norm_w, (float*)d_out);
}
